// GaussianGenerator_29832842838356
// MI455X (gfx1250) — matrix operand order repaired and checked by reading, not compiled, not run on hardware
//
#include <hip/hip_runtime.h>
#include <hip/hip_bf16.h>
#include <math.h>

// ---------------------------------------------------------------------------
// GMMConv (MoNet) 3-stage pipeline for MI455X (gfx1250, wave32, WMMA).
//
// Dense GEMMs (x@g, x@root, MLP layers) use v_wmma_f32_16x16x32_f16 with an
// LDS-staged, f16-transposed B chunk shared across the block's 8 waves and
// branchless, clamped global loads (zero-fill via cndmask only in the
// one-shot K tail). Irregular Gaussian aggregation runs on VALU + LDS.
// ---------------------------------------------------------------------------

typedef __attribute__((ext_vector_type(16))) _Float16 v16h;
typedef __attribute__((ext_vector_type(8)))  float    v8f;

#define KGM 25  // Gaussian mixture kernel size

__device__ __forceinline__ float act_apply(float x, int act) {
    if (act == 1) return x > 0.0f ? x : expm1f(x);   // ELU (alpha=1)
    if (act == 2) return tanhf(x);
    return x;
}

// ---------------------------------------------------------------------------
// Row-major GEMM: C[M,N] = act(A[M,K] @ B[K,N] + bias)
// 256 threads = 8 waves; each wave owns a 16x16 tile, waves stacked along M
// (block covers 128 rows x 16 cols). Requires M % 16 == 0 (true for all
// calls; M is always a multiple of 128). K in chunks of 32 with a single
// guarded tail. B chunk staged in LDS (f16, transposed). Out-of-range B
// columns are clamped (not zeroed): they only feed dead output columns.
// ---------------------------------------------------------------------------
__global__ __launch_bounds__(256)
void wmma_gemm_kernel(const float* __restrict__ A, int lda,
                      const float* __restrict__ Bw, int ldb,
                      float* __restrict__ C, int ldc,
                      int M, int N, int Kdim,
                      const float* __restrict__ bias, int act)
{
    __shared__ alignas(32) _Float16 sBt[16 * 32];   // [col][k] f16 B chunk

    const int tid  = threadIdx.x;
    const int wave = tid >> 5;
    const int lane = tid & 31;
    const int half = lane >> 4;      // lane group: 0 = lanes 0-15, 1 = 16-31
    const int lrow = lane & 15;

    const int tileM = (blockIdx.x * 8 + wave) * 16;
    const int tileN = blockIdx.y * 16;
    const bool mOK  = (tileM < M);   // wave-uniform

    // B staging coordinates (all 256 threads cooperate; covers 32x16 chunk).
    // Column clamped to N-1: out-of-range columns feed only dead lanes.
    const int bc  = tid & 15;        // column within tile
    const int br  = tid >> 4;        // k-row 0..15 (passes +0 and +16)
    const int bcc = min(tileN + bc, N - 1);
    const float* Bp0 = Bw + (unsigned)br * (unsigned)ldb + (unsigned)bcc;
    const unsigned bstep = (unsigned)ldb * 32u;   // elements per 32-row chunk

    const int arow = tileM + lrow;
    const float* Arow = A + (size_t)arow * lda;

    const int kfull = Kdim & ~31;
    v8f acc = {};

    for (int k0 = 0; k0 < kfull; k0 += 32) {
        __syncthreads();
        {   // stage B chunk -> LDS transposed, f16 (unconditional loads)
            float b0 = Bp0[0];
            float b1 = Bp0[(unsigned)ldb * 16u];
            sBt[bc * 32 + br + (br & 8)]     = (_Float16)b0;
            sBt[bc * 32 + br + 8 + (br & 8)] = (_Float16)b1;
            Bp0 += bstep;
        }
        __syncthreads();

        if (mOK) {
            v16h a;
            // A fragment: lane-half 0 -> K {0..7,16..23}, half 1 -> {8..15,24..31}
            const float* pa0 = Arow + k0 + half * 8;       // 8 contiguous
            const float* pa1 = pa0 + 16;                   // 8 contiguous
            #pragma unroll
            for (int j = 0; j < 4; ++j) {
                float2 u = *(const float2*)(pa0 + 2 * j);
                a[2 * j]     = (_Float16)u.x;
                a[2 * j + 1] = (_Float16)u.y;
                float2 v = *(const float2*)(pa1 + 2 * j);
                a[8 + 2 * j]     = (_Float16)v.x;
                a[8 + 2 * j + 1] = (_Float16)v.y;
            }
            // B fragment: 16 contiguous f16 in LDS (32B aligned)
            v16h b = *(const v16h*)&sBt[lrow * 32 + half * 16];
            acc = __builtin_amdgcn_wmma_f32_16x16x32_f16(
                false, a, false, b, (short)0, acc, false, false);
        }
    }

    if (kfull < Kdim) {   // K tail: executed at most once
        const int k0 = kfull;
        __syncthreads();
        {   // clamped unconditional loads + select-to-zero (k rows beyond
            // Kdim must contribute 0 since they are summed over)
            int kk0 = k0 + br, kk1 = k0 + br + 16;
            int ck0 = min(kk0, Kdim - 1), ck1 = min(kk1, Kdim - 1);
            float b0 = Bw[(unsigned)ck0 * (unsigned)ldb + (unsigned)bcc];
            float b1 = Bw[(unsigned)ck1 * (unsigned)ldb + (unsigned)bcc];
            b0 = (kk0 < Kdim) ? b0 : 0.0f;
            b1 = (kk1 < Kdim) ? b1 : 0.0f;
            sBt[bc * 32 + br + (br & 8)]     = (_Float16)b0;
            sBt[bc * 32 + br + 8 + (br & 8)] = (_Float16)b1;
        }
        __syncthreads();

        if (mOK) {
            v16h a;
            #pragma unroll
            for (int i = 0; i < 16; ++i) {
                int kk = k0 + ((i < 8) ? (half * 8 + i)
                                       : (16 + half * 8 + (i - 8)));
                float av = Arow[min(kk, Kdim - 1)];        // unconditional
                a[i] = (_Float16)((kk < Kdim) ? av : 0.0f);
            }
            v16h b = *(const v16h*)&sBt[lrow * 32 + half * 16];
            acc = __builtin_amdgcn_wmma_f32_16x16x32_f16(
                false, a, false, b, (short)0, acc, false, false);
        }
    }

    // Epilogue: VGPR v -> row tileM + half*8 + v, col tileN + lrow
    if (mOK) {
        int col = tileN + lrow;
        if (col < N) {
            float bv = bias ? bias[col] : 0.0f;
            #pragma unroll
            for (int v = 0; v < 8; ++v) {
                unsigned row = (unsigned)(tileM + half * 8 + v);
                C[row * (unsigned)ldc + (unsigned)col] = act_apply(acc[v] + bv, act);
            }
        }
    }
}

// ---------------------------------------------------------------------------
// p = concat([repeat(pos, rep, axis=0), x], axis=1)   out: [rows, 2+nx]
// ---------------------------------------------------------------------------
__global__ __launch_bounds__(256)
void concat_kernel(const float* __restrict__ pos, const float* __restrict__ x,
                   float* __restrict__ out, int rep, int nx, int rows)
{
    int i = blockIdx.x * blockDim.x + threadIdx.x;
    int w = 2 + nx;
    int total = rows * w;
    if (i >= total) return;
    int r = i / w, c = i % w;
    out[i] = (c < 2) ? pos[(unsigned)(r / rep) * 2u + (unsigned)c]
                     : x[(unsigned)r * (unsigned)nx + (unsigned)(c - 2)];
}

// ---------------------------------------------------------------------------
// GMMConv aggregation. One block per (graph b, target node t).
//   w[s,k] = mask(s,t) * exp(-0.5 * sum_d (pseudo_d - mu[k,d])^2/(sig^2+eps))
//   out[t,o] = act( (sum_{s,k} w[s,k]*xm[s, k*O+o]) / max(deg,1)
//                   + rootout[t,o] + bias[o] )
// ---------------------------------------------------------------------------
__global__ __launch_bounds__(128)
void gmm_agg_kernel(const float* __restrict__ pos,      // [B*n, 2]
                    const float* __restrict__ xm,       // [B*n, K*O]
                    const float* __restrict__ rootout,  // [B*n, O]
                    const float* __restrict__ bias,     // [O]
                    const float* __restrict__ mu,       // [K,2]
                    const float* __restrict__ sigma,    // [K,2]
                    float* __restrict__ out,            // [B*n, O]
                    int n, int O, int act)
{
    const int bt  = blockIdx.x;   // b*n + t
    const int b   = bt / n;
    const int t   = bt % n;
    const int tid = threadIdx.x;
    const int nth = blockDim.x;

    extern __shared__ float smem[];
    float* w    = smem;               // n*K
    float* sdeg = w + n * KGM;        // 1
    float* smu  = sdeg + 1;           // 2*K
    float* sisv = smu + 2 * KGM;      // 2*K  (1/(sigma^2+eps))
    float* red  = sisv + 2 * KGM;     // nth

    for (int i = tid; i < 2 * KGM; i += nth) {
        smu[i] = mu[i];
        float s = sigma[i];
        sisv[i] = 1.0f / (s * s + 1e-15f);
    }
    if (tid == 0) sdeg[0] = 0.0f;
    __syncthreads();

    const float tx = pos[(unsigned)bt * 2u + 0u];
    const float ty = pos[(unsigned)bt * 2u + 1u];

    for (int sk = tid; sk < n * KGM; sk += nth) {
        int s = sk / KGM, k = sk % KGM;
        float sx = pos[(unsigned)(b * n + s) * 2u + 0u];
        float sy = pos[(unsigned)(b * n + s) * 2u + 1u];
        float ddx = tx - sx;                     // diff = pos_t - pos_s
        float ddy = ty - sy;
        float ex = ddx + 1e-12f, ey = ddy + 1e-12f;
        float dist = sqrtf(ex * ex + ey * ey);
        bool  m = (dist < 0.5f) && (s != t);
        float px = ddx * 2.0f + 0.5f;            // diff/CUTOFF + 0.5
        float py = ddy * 2.0f + 0.5f;
        float qx = px - smu[k * 2 + 0];
        float qy = py - smu[k * 2 + 1];
        float e  = -0.5f * (qx * qx * sisv[k * 2 + 0] + qy * qy * sisv[k * 2 + 1]);
        w[sk] = m ? expf(e) : 0.0f;
        if (k == 0 && m) atomicAdd(sdeg, 1.0f);
    }
    __syncthreads();

    const float invdeg = 1.0f / fmaxf(sdeg[0], 1.0f);

    if (O > 1) {
        for (int o = tid; o < O; o += nth) {
            float acc = 0.0f;
            for (int s = 0; s < n; ++s) {
                const float* xr = xm + (unsigned)(b * n + s) * (unsigned)(KGM * O) + (unsigned)o;
                const float* wr = w + s * KGM;
                #pragma unroll 5
                for (int k = 0; k < KGM; ++k)
                    acc += wr[k] * xr[(unsigned)k * (unsigned)O];
            }
            float r = acc * invdeg + rootout[(unsigned)bt * (unsigned)O + (unsigned)o] + bias[o];
            out[(unsigned)bt * (unsigned)O + (unsigned)o] = act_apply(r, act);
        }
    } else {
        // O == 1: xm row stride = KGM, so xm index = (b*n)*KGM + sk
        float part = 0.0f;
        const float* xg = xm + (unsigned)(b * n) * (unsigned)KGM;
        for (int sk = tid; sk < n * KGM; sk += nth)
            part += w[sk] * xg[sk];
        red[tid] = part;
        __syncthreads();
        for (int st = nth >> 1; st > 0; st >>= 1) {
            if (tid < st) red[tid] += red[tid + st];
            __syncthreads();
        }
        if (tid == 0) {
            float r = red[0] * invdeg + rootout[bt] + bias[0];
            out[bt] = act_apply(r, act);
        }
    }
}

// ---------------------------------------------------------------------------
// Host-side launch
// ---------------------------------------------------------------------------
static inline void gemm(const float* A, int lda, const float* B, int ldb,
                        float* C, int ldc, int M, int N, int K,
                        const float* bias, int act, hipStream_t s)
{
    dim3 grid((M + 127) / 128, (N + 15) / 16);
    wmma_gemm_kernel<<<grid, 256, 0, s>>>(A, lda, B, ldb, C, ldc, M, N, K, bias, act);
}

static inline size_t agg_smem(int n) {
    return (size_t)(n * KGM + 1 + 4 * KGM + 128) * sizeof(float);
}

extern "C" void kernel_launch(void* const* d_in, const int* in_sizes, int n_in,
                              void* d_out, int out_size, void* d_ws, size_t ws_size,
                              hipStream_t stream) {
    (void)in_sizes; (void)n_in; (void)out_size; (void)ws_size;

    const float* x     = (const float*)d_in[0];   // [5120, 64]
    const float* pos   = (const float*)d_in[1];   // [5120, 2]
    const float* g1    = (const float*)d_in[2];   // [64, 4000]
    const float* mu1   = (const float*)d_in[3];
    const float* sig1  = (const float*)d_in[4];
    const float* root1 = (const float*)d_in[5];   // [64, 160]
    const float* cb1   = (const float*)d_in[6];
    const float* g2    = (const float*)d_in[7];   // [32, 1200]
    const float* mu2   = (const float*)d_in[8];
    const float* sig2  = (const float*)d_in[9];
    const float* root2 = (const float*)d_in[10];  // [32, 48]
    const float* cb2   = (const float*)d_in[11];
    const float* g3    = (const float*)d_in[12];  // [16, 25]
    const float* mu3   = (const float*)d_in[13];
    const float* sig3  = (const float*)d_in[14];
    const float* root3 = (const float*)d_in[15];  // [16, 1]
    const float* cb3   = (const float*)d_in[16];
    const float* p1w1  = (const float*)d_in[17];  // [34, 128]
    const float* p1b1  = (const float*)d_in[18];
    const float* p1w2  = (const float*)d_in[19];  // [128, 64]
    const float* p1b2  = (const float*)d_in[20];
    const float* p1w3  = (const float*)d_in[21];  // [64, 2]
    const float* p1b3  = (const float*)d_in[22];
    const float* p2w1  = (const float*)d_in[23];  // [18, 128]
    const float* p2b1  = (const float*)d_in[24];
    const float* p2w2  = (const float*)d_in[25];  // [128, 64]
    const float* p2b2  = (const float*)d_in[26];
    const float* p2w3  = (const float*)d_in[27];  // [64, 2]
    const float* p2b3  = (const float*)d_in[28];

    float* out  = (float*)d_out;          // [76800] x, then [76800*2] pos
    float* xout = out;
    float* pos3 = out + 76800;

    // Workspace slots (floats), lifetimes disjoint where aliased.
    float* ws   = (float*)d_ws;
    float* xmS  = ws;                     // 30,720,000 (xm1 / xm2 / xm3)
    float* rS   = xmS + 30720000;         //  1,228,800 (r1 / r2 / r3)
    float* h1   = rS  + 1228800;          //    819,200 ([5120,160] == [25600,32])
    float* pos2 = h1  + 819200;           //     51,200 ([25600,2])
    float* h2   = pos2 + 51200;           //  1,228,800 ([25600,48] == [76800,16])
    float* pin  = h2  + 1228800;          //  1,382,400 (p1in / p2in)
    float* t1   = pin + 1382400;          //  9,830,400
    float* t2   = t1  + 9830400;          //  4,915,200

    // ---------------- Stage 1 (n=5, O=160) ----------------
    gemm(x, 64, g1,    4000, xmS, 4000, 5120, 4000, 64, nullptr, 0, stream);
    gemm(x, 64, root1,  160, rS,   160, 5120,  160, 64, nullptr, 0, stream);
    gmm_agg_kernel<<<5120, 128, agg_smem(5), stream>>>(
        pos, xmS, rS, cb1, mu1, sig1, h1, 5, 160, /*ELU*/1);

    // p = [repeat(pos,5), h1 as (25600,32)] -> MLP 34->128->64->2 (ELU each)
    {
        int total = 25600 * 34;
        concat_kernel<<<(total + 255) / 256, 256, 0, stream>>>(pos, h1, pin, 5, 32, 25600);
    }
    gemm(pin, 34, p1w1, 128, t1,  128, 25600, 128,  34, p1b1, 1, stream);
    gemm(t1, 128, p1w2,  64, t2,   64, 25600,  64, 128, p1b2, 1, stream);
    gemm(t2,  64, p1w3,   2, pos2,  2, 25600,   2,  64, p1b3, 1, stream);

    // ---------------- Stage 2 (n=25, O=48) ----------------
    gemm(h1, 32, g2,    1200, xmS, 1200, 25600, 1200, 32, nullptr, 0, stream);
    gemm(h1, 32, root2,   48, rS,    48, 25600,   48, 32, nullptr, 0, stream);
    gmm_agg_kernel<<<25600, 128, agg_smem(25), stream>>>(
        pos2, xmS, rS, cb2, mu2, sig2, h2, 25, 48, /*ELU*/1);

    // p = [repeat(pos2,3), h2 as (76800,16)] -> MLP 18->128->64->2 (ELU each)
    {
        int total = 76800 * 18;
        concat_kernel<<<(total + 255) / 256, 256, 0, stream>>>(pos2, h2, pin, 3, 16, 76800);
    }
    gemm(pin, 18, p2w1, 128, t1,  128, 76800, 128,  18, p2b1, 1, stream);
    gemm(t1, 128, p2w2,  64, t2,   64, 76800,  64, 128, p2b2, 1, stream);
    gemm(t2,  64, p2w3,   2, pos3,  2, 76800,   2,  64, p2b3, 1, stream);  // pos out

    // ---------------- Stage 3 (n=75, O=1, tanh) ----------------
    gemm(h2, 16, g3,    25, xmS, 25, 76800, 25, 16, nullptr, 0, stream);
    gemm(h2, 16, root3,  1, rS,   1, 76800,  1, 16, nullptr, 0, stream);
    gmm_agg_kernel<<<76800, 128, agg_smem(75), stream>>>(
        pos3, xmS, rS, cb3, mu3, sig3, xout, 75, 1, /*tanh*/2);
}
